// GraphRNNEncoder_66924180407101
// MI455X (gfx1250) — matrix operand order repaired and checked by reading, not compiled, not run on hardware
//
#include <hip/hip_runtime.h>
#include <hip/hip_bf16.h>

// ---------------------------------------------------------------------------
// GraphRNN encoder for MI455X (gfx1250, wave32, WMMA).
// State m kept in bf16 (134MB -> L2-resident), all GEMMs via
// v_wmma_f32_16x16x32_bf16 with f32 accumulation.
// ---------------------------------------------------------------------------

#define TT  4
#define NN  256
#define FF  64
#define OUTN 512
#define BB  16

typedef __bf16 bf16;
typedef bf16  v16bf __attribute__((ext_vector_type(16)));
typedef bf16  v8bf  __attribute__((ext_vector_type(8)));
typedef float v8f   __attribute__((ext_vector_type(8)));

__device__ __forceinline__ v8f wmma_bf16(v16bf a, v16bf b, v8f c) {
  // (neg_a, A, neg_b, B, c_mod, C, reuse_a, reuse_b)
  return __builtin_amdgcn_wmma_f32_16x16x32_bf16(false, a, false, b, (short)0, c,
                                                 false, false);
}

// A fragment (16x32 bf16, row major): lane supplies its row pointer (row = lane&15).
// Lanes 0-15 hold K = 0..15, lanes 16-31 hold K = 16..31 (matches B fragment).
__device__ __forceinline__ v16bf load_a_frag(const bf16* __restrict__ rowptr, int kb) {
  const int off = ((threadIdx.x >> 4) & 1) << 4;  // 0 or 16
  return *(const v16bf*)(rowptr + kb + off);
}

// B fragment (32x16): B[k][n] = W[n0+n][kb+k] (row-major W => computes X @ W^T).
// Lane layout: n = lane&15, K block = (lane>>4)*16 + e  -> one 32B load.
__device__ __forceinline__ v16bf load_b_frag(const bf16* __restrict__ W, int ldw,
                                             int n0, int kb) {
  const int lane = threadIdx.x & 31;
  const bf16* p = W + (size_t)(n0 + (lane & 15)) * (size_t)ldw + kb +
                  (((lane >> 4) & 1) << 4);
  return *(const v16bf*)p;
}

// ------------------------------- prep kernels ------------------------------

__global__ void k_f32_to_bf16(const float* __restrict__ src, bf16* __restrict__ dst,
                              int n) {
  int i = blockIdx.x * blockDim.x + threadIdx.x;
  if (i < n) dst[i] = (bf16)src[i];
}

// xT[b][g][n] = x[b][n][g]  (so x can serve as a WMMA B operand for U @ x)
__global__ void k_xT(const float* __restrict__ x, bf16* __restrict__ xT) {
  int i = blockIdx.x * blockDim.x + threadIdx.x;
  if (i >= BB * FF * NN) return;
  int n = i % NN;
  int g = (i / NN) % FF;
  int b = i / (NN * FF);
  xT[i] = (bf16)x[((size_t)b * NN + n) * FF + g];
}

// ------------------------- generic 16-row x 64 GEMMs -----------------------

// Base[r,:] = X[r,:] @ W^T   (rows of 64, K=64), one wave per 16-row tile
__global__ void k_base(const bf16* __restrict__ X, const bf16* __restrict__ W,
                       float* __restrict__ Out) {
  const int r0 = blockIdx.x * 16;
  const int l15 = threadIdx.x & 15, mb = ((threadIdx.x >> 4) & 1) << 3;
  const bf16* rowptr = X + (size_t)(r0 + l15) * FF;
  v16bf a0 = load_a_frag(rowptr, 0);
  v16bf a1 = load_a_frag(rowptr, 32);
#pragma unroll
  for (int t = 0; t < 4; ++t) {
    v8f acc = {};
    acc = wmma_bf16(a0, load_b_frag(W, FF, t * 16, 0), acc);
    acc = wmma_bf16(a1, load_b_frag(W, FF, t * 16, 32), acc);
    const int n = t * 16 + l15;
#pragma unroll
    for (int r = 0; r < 8; ++r)
      Out[(size_t)(r0 + mb + r) * FF + n] = acc[r];
  }
}

// P[r,:] = Base[r,:] + S[r,:] @ Wnm^T
__global__ void k_P(const bf16* __restrict__ S, const float* __restrict__ Base,
                    const bf16* __restrict__ Wnm, float* __restrict__ P) {
  const int r0 = blockIdx.x * 16;
  const int l15 = threadIdx.x & 15, mb = ((threadIdx.x >> 4) & 1) << 3;
  const bf16* rowptr = S + (size_t)(r0 + l15) * FF;
  v16bf a0 = load_a_frag(rowptr, 0);
  v16bf a1 = load_a_frag(rowptr, 32);
#pragma unroll
  for (int t = 0; t < 4; ++t) {
    v8f acc = {};
    acc = wmma_bf16(a0, load_b_frag(Wnm, FF, t * 16, 0), acc);
    acc = wmma_bf16(a1, load_b_frag(Wnm, FF, t * 16, 32), acc);
    const int n = t * 16 + l15;
#pragma unroll
    for (int r = 0; r < 8; ++r) {
      size_t idx = (size_t)(r0 + mb + r) * FF + n;
      P[idx] = Base[idx] + acc[r];
    }
  }
}

// ------------------------------- reductions --------------------------------

// S[b,d,f] = sum_s mT[b,d,s,f]   (contiguous rows; streaming, prefetched)
__global__ void k_reduce_rows(const bf16* __restrict__ M, bf16* __restrict__ Sout) {
  const int bd = blockIdx.x;          // b*NN + d
  const int f = threadIdx.x;          // 0..63
  const bf16* p = M + (size_t)bd * NN * FF + f;
  float acc = 0.f;
  for (int s = 0; s < NN; ++s) {
    if ((s & 7) == 0) __builtin_prefetch(p + (size_t)(s + 16) * FF, 0, 0);
    acc += (float)p[(size_t)s * FF];
  }
  Sout[(size_t)bd * FF + f] = (bf16)acc;
}

// summed[b,i,f] = sum_d mT[b,d,i,f]   (final pass, strided over d)
__global__ void k_reduce_cols(const bf16* __restrict__ M, bf16* __restrict__ Sm) {
  const int b = blockIdx.x / NN;
  const int i = blockIdx.x % NN;
  const int f = threadIdx.x;
  const bf16* p = M + (size_t)b * NN * NN * FF + (size_t)i * FF + f;
  float acc = 0.f;
  for (int d = 0; d < NN; ++d) acc += (float)p[(size_t)d * NN * FF];
  Sm[(size_t)blockIdx.x * FF + f] = (bf16)acc;
}

// -------------------------------- step t=0 ---------------------------------

// mT1[b,d,s,g] = A[b,s,d] * max(0, base[b,s,g])   (m starts at zero => S=0)
__global__ void k_init_m(const float* __restrict__ adj, const float* __restrict__ Base,
                         bf16* __restrict__ Mout) {
  const int bd = blockIdx.x;  // b*NN + d
  const int b = bd / NN, d = bd % NN;
  bf16* out = Mout + (size_t)bd * NN * FF;
  for (int e = threadIdx.x; e < NN * FF; e += blockDim.x) {
    const int s = e >> 6, g = e & 63;
    const float a = adj[((size_t)b * NN + s) * NN + d];
    const float v = Base[((size_t)b * NN + s) * FF + g];
    out[e] = (bf16)(a * fmaxf(v, 0.f));
  }
}

// ------------------------------ main step kernel ---------------------------

// new_mT[b,d,s,:] = A[b,s,d] * relu(P[b,s,:] - mT_old[b,s,d,:] @ Wnm^T)
// One wave per (b, d, 16-wide s-tile). Output staged in LDS for coalesced
// 128-bit bf16 stores of the contiguous [16,64] destination block.
__global__ void k_update(const bf16* __restrict__ Mold, const float* __restrict__ P,
                         const float* __restrict__ adj, const bf16* __restrict__ Wnm,
                         bf16* __restrict__ Mnew) {
  __shared__ float tile[16 * 68];       // 68-stride pad: disjoint LDS banks
  const int st = blockIdx.x & 15;
  const int d = (blockIdx.x >> 4) & (NN - 1);
  const int b = blockIdx.x >> 12;
  const int s0 = st * 16;
  const int lane = threadIdx.x;
  const int l15 = lane & 15;
  const int mb = (lane >> 4) << 3;      // 0 or 8

  // A rows: mT_old[b, s0+m, d, :]  (each row 64 contiguous bf16)
  const bf16* rowptr = Mold + (((size_t)b * NN + (s0 + l15)) * NN + d) * FF;
  v16bf a0 = load_a_frag(rowptr, 0);
  v16bf a1 = load_a_frag(rowptr, 32);

  float av[8];
#pragma unroll
  for (int r = 0; r < 8; ++r)
    av[r] = adj[((size_t)b * NN + (s0 + mb + r)) * NN + d];

#pragma unroll
  for (int t = 0; t < 4; ++t) {
    v8f acc = {};
    acc = wmma_bf16(a0, load_b_frag(Wnm, FF, t * 16, 0), acc);
    acc = wmma_bf16(a1, load_b_frag(Wnm, FF, t * 16, 32), acc);
    const int n = t * 16 + l15;
#pragma unroll
    for (int r = 0; r < 8; ++r) {
      const int s = s0 + mb + r;
      const float p = P[((size_t)b * NN + s) * FF + n];
      tile[(mb + r) * 68 + n] = av[r] * fmaxf(p - acc[r], 0.f);
    }
  }
  __syncthreads();

  // lane -> (row = lane&15, half = lane>>4): 4 x 16B contiguous bf16 stores
  bf16* dst = Mnew + (((size_t)b * NN + d) * NN + (s0 + l15)) * FF + ((lane >> 4) << 5);
  const float* srcrow = &tile[l15 * 68 + ((lane >> 4) << 5)];
#pragma unroll
  for (int j = 0; j < 4; ++j) {
    v8bf v;
#pragma unroll
    for (int i = 0; i < 8; ++i) v[i] = (bf16)srcrow[j * 8 + i];
    *(v8bf*)(dst + j * 8) = v;
  }
}

// --------------------------------- encoder ---------------------------------

// EncF[b,i,g] = sum_n u_nf[i,n] * x[b,n,g]   (A = U rows, B = x^T rows, K=256)
__global__ void k_ux(const bf16* __restrict__ U, const bf16* __restrict__ xT,
                     float* __restrict__ EncF) {
  const int it = blockIdx.x & 15;
  const int b = blockIdx.x >> 4;
  const int i0 = it * 16;
  const int l15 = threadIdx.x & 15, mb = ((threadIdx.x >> 4) & 1) << 3;
  const bf16* urow = U + (size_t)(i0 + l15) * NN;
  const bf16* xTb = xT + (size_t)b * FF * NN;
  v8f z = {};
  v8f acc[4] = {z, z, z, z};
  for (int kb = 0; kb < NN; kb += 32) {
    v16bf a = load_a_frag(urow, kb);
#pragma unroll
    for (int t = 0; t < 4; ++t)
      acc[t] = wmma_bf16(a, load_b_frag(xTb, NN, t * 16, kb), acc[t]);
  }
#pragma unroll
  for (int t = 0; t < 4; ++t) {
    const int n = t * 16 + l15;
#pragma unroll
    for (int r = 0; r < 8; ++r)
      EncF[((size_t)b * NN + i0 + mb + r) * FF + n] = acc[t][r];
  }
}

// EncBF = bf16(relu(EncF + summed @ u_nm^T))
__global__ void k_smt(const bf16* __restrict__ Sm, const bf16* __restrict__ Unm,
                      const float* __restrict__ EncF, bf16* __restrict__ EncBF) {
  const int r0 = blockIdx.x * 16;
  const int l15 = threadIdx.x & 15, mb = ((threadIdx.x >> 4) & 1) << 3;
  const bf16* rowptr = Sm + (size_t)(r0 + l15) * FF;
  v16bf a0 = load_a_frag(rowptr, 0);
  v16bf a1 = load_a_frag(rowptr, 32);
#pragma unroll
  for (int t = 0; t < 4; ++t) {
    v8f acc = {};
    acc = wmma_bf16(a0, load_b_frag(Unm, FF, t * 16, 0), acc);
    acc = wmma_bf16(a1, load_b_frag(Unm, FF, t * 16, 32), acc);
    const int n = t * 16 + l15;
#pragma unroll
    for (int r = 0; r < 8; ++r) {
      size_t idx = (size_t)(r0 + mb + r) * FF + n;
      EncBF[idx] = (bf16)fmaxf(EncF[idx] + acc[r], 0.f);
    }
  }
}

// out[b,o] = sigmoid(enc_flat[b,:] . lw[o,:] + lb[o])
// M=16 == batch fits one WMMA tile; K=16384 split over 8 waves + LDS reduce.
__global__ void k_final(const bf16* __restrict__ Enc, const bf16* __restrict__ LW,
                        const float* __restrict__ lb, float* __restrict__ out) {
  __shared__ float red[8][32][8];
  const int o0 = blockIdx.x * 16;
  const int wv = threadIdx.x >> 5;
  const int lane = threadIdx.x & 31;
  const int l15 = lane & 15, mb = (lane >> 4) << 3;
  const int K = NN * FF;  // 16384
  const bf16* erow = Enc + (size_t)l15 * K;  // A row = batch index
  v8f acc = {};
  for (int kb = wv * 32; kb < K; kb += 8 * 32) {
    v16bf a = load_a_frag(erow, kb);
    v16bf bf = load_b_frag(LW, K, o0, kb);
    acc = wmma_bf16(a, bf, acc);
  }
#pragma unroll
  for (int r = 0; r < 8; ++r) red[wv][lane][r] = acc[r];
  __syncthreads();
  if (wv == 0) {
#pragma unroll
    for (int r = 0; r < 8; ++r) {
      float s = 0.f;
#pragma unroll
      for (int w = 0; w < 8; ++w) s += red[w][lane][r];
      const int bidx = mb + r;
      const int o = o0 + l15;
      s += lb[o];
      out[(size_t)bidx * OUTN + o] = 1.f / (1.f + __expf(-s));
    }
  }
}

// --------------------------------- launcher --------------------------------

extern "C" void kernel_launch(void* const* d_in, const int* in_sizes, int n_in,
                              void* d_out, int out_size, void* d_ws, size_t ws_size,
                              hipStream_t stream) {
  (void)in_sizes; (void)n_in; (void)out_size; (void)ws_size;
  const float* x    = (const float*)d_in[0];
  const float* adj  = (const float*)d_in[1];
  const float* w_nf = (const float*)d_in[2];
  const float* w_nm = (const float*)d_in[3];
  const float* u_nf = (const float*)d_in[4];
  const float* u_nm = (const float*)d_in[5];
  const float* lw   = (const float*)d_in[6];
  const float* lb   = (const float*)d_in[7];
  float* out = (float*)d_out;

  char* ws = (char*)d_ws;
  const size_t SM = (size_t)BB * NN * NN * FF * 2;  // 134 MB bf16 state
  bf16* M0 = (bf16*)ws;
  bf16* M1 = (bf16*)(ws + SM);
  char* p = ws + 2 * SM;
  float* Base = (float*)p;  p += (size_t)BB * NN * FF * 4;
  float* P    = (float*)p;  p += (size_t)BB * NN * FF * 4;
  bf16* S     = (bf16*)p;   p += (size_t)BB * NN * FF * 2;
  bf16* Sm    = (bf16*)p;   p += (size_t)BB * NN * FF * 2;
  float* EncF = (float*)p;  p += (size_t)BB * NN * FF * 4;
  bf16* EncBF = (bf16*)p;   p += (size_t)BB * NN * FF * 2;
  bf16* Xbf   = (bf16*)p;   p += (size_t)BB * NN * FF * 2;
  bf16* XTbf  = (bf16*)p;   p += (size_t)BB * FF * NN * 2;
  bf16* Wnf   = (bf16*)p;   p += (size_t)FF * FF * 2;
  bf16* Wnm   = (bf16*)p;   p += (size_t)FF * FF * 2;
  bf16* Unm   = (bf16*)p;   p += (size_t)FF * FF * 2;
  bf16* Unf   = (bf16*)p;   p += (size_t)NN * NN * 2;
  bf16* LWbf  = (bf16*)p;   p += (size_t)OUTN * NN * FF * 2;

  auto cvt = [&](const float* s, bf16* d, int n) {
    k_f32_to_bf16<<<(n + 255) / 256, 256, 0, stream>>>(s, d, n);
  };
  cvt(x, Xbf, BB * NN * FF);
  cvt(w_nf, Wnf, FF * FF);
  cvt(w_nm, Wnm, FF * FF);
  cvt(u_nm, Unm, FF * FF);
  cvt(u_nf, Unf, NN * NN);
  cvt(lw, LWbf, OUTN * NN * FF);
  k_xT<<<(BB * FF * NN + 255) / 256, 256, 0, stream>>>(x, XTbf);

  // base[b,n,:] = x[b,n,:] @ w_nf^T
  k_base<<<BB * NN / 16, 32, 0, stream>>>(Xbf, Wnf, Base);
  // step 1 (m0 == 0): m1 = A * relu(base)
  k_init_m<<<BB * NN, 256, 0, stream>>>(adj, Base, M0);

  bf16* cur = M0;
  bf16* nxt = M1;
  for (int t = 1; t < TT; ++t) {
    k_reduce_rows<<<BB * NN, 64, 0, stream>>>(cur, S);
    k_P<<<BB * NN / 16, 32, 0, stream>>>(S, Base, Wnm, P);
    k_update<<<BB * NN * (NN / 16), 32, 0, stream>>>(cur, P, adj, Wnm, nxt);
    bf16* tmp = cur; cur = nxt; nxt = tmp;
  }

  k_reduce_cols<<<BB * NN, 64, 0, stream>>>(cur, Sm);
  k_ux<<<BB * 16, 32, 0, stream>>>(Unf, XTbf, EncF);
  k_smt<<<BB * NN / 16, 32, 0, stream>>>(Sm, Unm, EncF, EncBF);
  k_final<<<OUTN / 16, 256, 0, stream>>>(EncBF, LWbf, lb, out);
}
